// SpatialSelfAttention_24421184045550
// MI455X (gfx1250) — matrix operand order repaired and checked by reading, not compiled, not run on hardware
//
#include <hip/hip_runtime.h>
#include <hip/hip_bf16.h>

typedef _Float16 v4h  __attribute__((ext_vector_type(4)));
typedef _Float16 v8h  __attribute__((ext_vector_type(8)));
typedef _Float16 v16h __attribute__((ext_vector_type(16)));
typedef float    v8f  __attribute__((ext_vector_type(8)));

#define XN_LD 264   // halves per row of xn  (264*2 = 528 B, 16B aligned)
#define HD_LD 72    // halves per row of per-head tiles (144 B, 16B aligned)

// ---- WMMA fragment loaders -------------------------------------------------
// A fragment (16x32 f16, M x K): lane<16 -> row = lane&15, K = k0..k0+15
// contiguous; lane>=16 -> K = k0+16..k0+31.
__device__ __forceinline__ v16h load_a_frag(const _Float16* base, int stride,
                                            int row0, int k0, int lane) {
    int hh = lane >> 4;
    const _Float16* p = base + (row0 + (lane & 15)) * stride + k0 + hh * 16;
    union { v16h v; v8h h[2]; } u;
    u.h[0] = *(const v8h*)(p);
    u.h[1] = *(const v8h*)(p + 8);
    return u.v;
}

// B fragment (32x16 f16, K x N) where the 16 K-values per lane are CONTIGUOUS
// halves at row (col0 + lane&15): lane<16 -> K=k0..k0+15, lane>=16 -> +16.
__device__ __forceinline__ v16h load_b_frag_lds(const _Float16* base, int stride,
                                                int col0, int k0, int lane) {
    const _Float16* p = base + (col0 + (lane & 15)) * stride + k0 + ((lane >> 4) << 4);
    union { v16h v; v8h h[2]; } u;
    u.h[0] = *(const v8h*)(p);
    u.h[1] = *(const v8h*)(p + 8);
    return u.v;
}

// B fragment from pre-packed f16 weights Wt[col][k] (ld = 256): per lane two
// aligned 16B global loads, zero conversion work. Weights stay L2-resident.
__device__ __forceinline__ v16h load_b_frag_packed(const _Float16* __restrict__ Wt,
                                                   int col0, int k0, int lane) {
    const _Float16* p = Wt + (size_t)(col0 + (lane & 15)) * 256 + k0 + ((lane >> 4) << 4);
    union { v16h v; v8h h[2]; } u;
    u.h[0] = *(const v8h*)(p);
    u.h[1] = *(const v8h*)(p + 8);
    return u.v;
}

// Fallback: gather from row-major f32 weight matrix W[k][n] (ld = ldw).
__device__ __forceinline__ v16h load_b_frag_w(const float* __restrict__ W, int ldw,
                                              int col0, int k0, int lane) {
    int col = col0 + (lane & 15);
    int kr  = k0 + ((lane >> 4) << 4);
    union { v16h v; _Float16 e[16]; } u;
#pragma unroll
    for (int i = 0; i < 16; ++i)
        u.e[i] = (_Float16)W[(kr + i) * ldw + col];
    return u.v;
}

#define WMMA_F16(A, B, C) \
    __builtin_amdgcn_wmma_f32_16x16x32_f16(false, (A), false, (B), (short)0, (C), false, false)

// ---- weight pre-pack: Wt[mat][col][k] = (f16) W_mat[k][col] ----------------
__global__ __launch_bounds__(256)
void pack_weights_kernel(const float* __restrict__ Wq, const float* __restrict__ Wk,
                         const float* __restrict__ Wv, const float* __restrict__ Wo,
                         _Float16* __restrict__ Wt) {
    const int mat = blockIdx.y;            // 0=Q 1=K 2=V 3=O
    const int col = blockIdx.x;            // 0..255
    const int k   = threadIdx.x;           // 0..255
    const float* W = (mat == 0) ? Wq : (mat == 1) ? Wk : (mat == 2) ? Wv : Wo;
    Wt[((size_t)mat * 256 + col) * 256 + k] = (_Float16)W[k * 256 + col];
}

template <bool PACKED>
__global__ __launch_bounds__(256)
void win_attn_kernel(const float* __restrict__ x,
                     const float* __restrict__ ln_gamma,
                     const float* __restrict__ ln_beta,
                     const float* __restrict__ Wq, const float* __restrict__ bq,
                     const float* __restrict__ Wk, const float* __restrict__ bk,
                     const float* __restrict__ Wv, const float* __restrict__ bv,
                     const float* __restrict__ Wo, const float* __restrict__ bo,
                     const _Float16* __restrict__ Wt,
                     float* __restrict__ out) {
    __shared__ float    sg[256];
    __shared__ float    sb[256];
    __shared__ _Float16 xnL[64 * XN_LD];   // normalized tokens, f16
    __shared__ _Float16 qhL[64 * HD_LD];   // per-head Q  [tok][d]  (aliased by attn)
    __shared__ _Float16 khL[64 * HD_LD];   // per-head K  [tok][d]  (aliased by ctx)
    __shared__ _Float16 vtL[64 * HD_LD];   // per-head V^T [d][tok]

    const int tid  = threadIdx.x;
    const int lane = tid & 31;
    // wave index is uniform per wave: pin it to an SGPR so wave-level branches
    // compile to scalar jumps (s_cbranch) instead of exec masking — WMMA needs
    // EXEC = all ones.
    const int wave = __builtin_amdgcn_readfirstlane(tid >> 5);

    const int wid = blockIdx.x;            // 0..4095
    const int bb  = wid >> 10;             // batch
    const int rem = wid & 1023;
    const int ht  = rem >> 5;              // window row  (0..31)
    const int wt  = rem & 31;              // window col  (0..31)

    auto rowbase = [&](int t) -> size_t {
        int r = t >> 3, cc = t & 7;
        return (((size_t)(bb * 256 + ht * 8 + r) * 256) + (size_t)(wt * 8 + cc)) * 256;
    };

    sg[tid] = ln_gamma[tid];
    sb[tid] = ln_beta[tid];

    // ---- LayerNorm (two-pass, 4 threads per token) -------------------------
    {
        const int t  = tid >> 2;
        const int p  = tid & 3;
        const size_t gb = rowbase(t) + p * 64;
        const float4* xp = (const float4*)(x + gb);
        float s1 = 0.f, s2 = 0.f;
#pragma unroll
        for (int i = 0; i < 16; ++i) {
            float4 v = xp[i];
            s1 += v.x + v.y + v.z + v.w;
            s2 += v.x * v.x + v.y * v.y + v.z * v.z + v.w * v.w;
        }
        s1 += __shfl_xor(s1, 1, 32); s1 += __shfl_xor(s1, 2, 32);
        s2 += __shfl_xor(s2, 1, 32); s2 += __shfl_xor(s2, 2, 32);
        const float mean = s1 * (1.0f / 256.0f);
        const float var  = s2 * (1.0f / 256.0f) - mean * mean;
        const float rstd = rsqrtf(var + 1e-5f);
        __syncthreads();                   // sg/sb visible
#pragma unroll
        for (int i = 0; i < 16; ++i) {
            float4 v = xp[i];
            int c = p * 64 + i * 4;
            v4h h;
            h.x = (_Float16)((v.x - mean) * rstd * sg[c + 0] + sb[c + 0]);
            h.y = (_Float16)((v.y - mean) * rstd * sg[c + 1] + sb[c + 1]);
            h.z = (_Float16)((v.z - mean) * rstd * sg[c + 2] + sb[c + 2]);
            h.w = (_Float16)((v.w - mean) * rstd * sg[c + 3] + sb[c + 3]);
            *(v4h*)(xnL + t * XN_LD + c) = h;
        }
    }
    __syncthreads();

    // persistent output accumulators: this wave owns c-tiles n = 2*wave, 2*wave+1
    v8f oacc[4][2];
#pragma unroll
    for (int m = 0; m < 4; ++m)
#pragma unroll
        for (int nn = 0; nn < 2; ++nn) oacc[m][nn] = (v8f)0.0f;

    const int hh = lane >> 4;

    for (int hd = 0; hd < 4; ++hd) {
        // ---- per-head Q/K/V projection: 12 column jobs over 8 waves --------
        for (int job = wave; job < 12; job += 8) {
            const int mat = job >> 2;      // 0=Q 1=K 2=V
            const int nt  = job & 3;       // n-tile within head (d dim)
            const float*    W    = (mat == 0) ? Wq : ((mat == 1) ? Wk : Wv);
            const float*    bias = (mat == 0) ? bq : ((mat == 1) ? bk : bv);
            const _Float16* WtM  = Wt + (size_t)mat * 256 * 256;
            const int col0 = hd * 64 + nt * 16;
            v8f acc[4];
#pragma unroll
            for (int m = 0; m < 4; ++m) acc[m] = (v8f)0.0f;
#pragma unroll 4
            for (int ks = 0; ks < 8; ++ks) {
                v16h bf = PACKED ? load_b_frag_packed(WtM, col0, ks * 32, lane)
                                 : load_b_frag_w(W, 256, col0, ks * 32, lane);
#pragma unroll
                for (int m = 0; m < 4; ++m) {
                    v16h af = load_a_frag(xnL, XN_LD, m * 16, ks * 32, lane);
                    acc[m] = WMMA_F16(af, bf, acc[m]);
                }
            }
            const float bvv  = bias[col0 + (lane & 15)];
            const int   cold = nt * 16 + (lane & 15);
#pragma unroll
            for (int m = 0; m < 4; ++m) {
#pragma unroll
                for (int j = 0; j < 8; ++j) {
                    float val = acc[m][j] + bvv;
                    int tok = m * 16 + j + hh * 8;
                    if (mat == 0)      qhL[tok * HD_LD + cold] = (_Float16)val;
                    else if (mat == 1) khL[tok * HD_LD + cold] = (_Float16)val;
                    else               vtL[cold * HD_LD + tok] = (_Float16)val; // V^T
                }
            }
        }
        __syncthreads();

        // ---- scores + softmax: wave m (<4) owns query rows [16m,16m+16) ----
        if (wave < 4) {
            const int m = wave;
            v8f s[4];
#pragma unroll
            for (int n = 0; n < 4; ++n) s[n] = (v8f)0.0f;
#pragma unroll
            for (int ks = 0; ks < 2; ++ks) {
                v16h af = load_a_frag(qhL, HD_LD, m * 16, ks * 32, lane);
#pragma unroll
                for (int n = 0; n < 4; ++n) {
                    v16h bf = load_b_frag_lds(khL, HD_LD, n * 16, ks * 32, lane);
                    s[n] = WMMA_F16(af, bf, s[n]);
                }
            }
            const float scale = 0.125f;    // 1/sqrt(64)
#pragma unroll
            for (int j = 0; j < 8; ++j) {
                float mx = -1e30f;
#pragma unroll
                for (int n = 0; n < 4; ++n) { s[n][j] *= scale; mx = fmaxf(mx, s[n][j]); }
                mx = fmaxf(mx, __shfl_xor(mx, 1, 32));
                mx = fmaxf(mx, __shfl_xor(mx, 2, 32));
                mx = fmaxf(mx, __shfl_xor(mx, 4, 32));
                mx = fmaxf(mx, __shfl_xor(mx, 8, 32));
                float sum = 0.f;
#pragma unroll
                for (int n = 0; n < 4; ++n) { float e = __expf(s[n][j] - mx); s[n][j] = e; sum += e; }
                sum += __shfl_xor(sum, 1, 32);
                sum += __shfl_xor(sum, 2, 32);
                sum += __shfl_xor(sum, 4, 32);
                sum += __shfl_xor(sum, 8, 32);
                const float inv = 1.0f / sum;
                const int tok = m * 16 + j + hh * 8;      // rows owned by this wave only
#pragma unroll
                for (int n = 0; n < 4; ++n)
                    qhL[tok * HD_LD + n * 16 + (lane & 15)] = (_Float16)(s[n][j] * inv);
            }
        }
        __syncthreads();

        // ---- ctx = attn @ V : 16 tiles, 2 per wave; write into khL ---------
#pragma unroll
        for (int q2 = 0; q2 < 2; ++q2) {
            const int id = wave * 2 + q2;
            const int m = id >> 2, n = id & 3;
            v8f c = (v8f)0.0f;
#pragma unroll
            for (int ks = 0; ks < 2; ++ks) {
                v16h af = load_a_frag(qhL, HD_LD, m * 16, ks * 32, lane);        // attn
                v16h bf = load_b_frag_lds(vtL, HD_LD, n * 16, ks * 32, lane);    // V^T rows
                c = WMMA_F16(af, bf, c);
            }
#pragma unroll
            for (int j = 0; j < 8; ++j) {
                int tok = m * 16 + j + hh * 8;
                khL[tok * HD_LD + n * 16 + (lane & 15)] = (_Float16)c[j];        // ctx[t][d]
            }
        }
        __syncthreads();

        // ---- out += ctx @ Wo_h : persistent accumulators -------------------
        const float*    WoH  = Wo + (size_t)hd * 64 * 256;
        const _Float16* WtO  = Wt + (size_t)3 * 256 * 256;     // Wt[3][c][h*64+d]
#pragma unroll
        for (int nn = 0; nn < 2; ++nn) {
            const int n = wave * 2 + nn;   // c-tile 0..15
#pragma unroll
            for (int ks = 0; ks < 2; ++ks) {
                v16h bf = PACKED ? load_b_frag_packed(WtO, n * 16, hd * 64 + ks * 32, lane)
                                 : load_b_frag_w(WoH, 256, n * 16, ks * 32, lane);
#pragma unroll
                for (int m = 0; m < 4; ++m) {
                    v16h af = load_a_frag(khL, HD_LD, m * 16, ks * 32, lane);    // ctx
                    oacc[m][nn] = WMMA_F16(af, bf, oacc[m][nn]);
                }
            }
        }
        __syncthreads();                   // khL/qhL/vtL reused next head
    }

    // ---- epilogue: y = x + out + bo ---------------------------------------
#pragma unroll
    for (int m = 0; m < 4; ++m) {
#pragma unroll
        for (int nn = 0; nn < 2; ++nn) {
            const int n = wave * 2 + nn;
            const int c = n * 16 + (lane & 15);
            const float bov = bo[c];
#pragma unroll
            for (int j = 0; j < 8; ++j) {
                const int tok = m * 16 + j + hh * 8;
                const size_t a = rowbase(tok) + c;
                out[a] = x[a] + oacc[m][nn][j] + bov;
            }
        }
    }
}

extern "C" void kernel_launch(void* const* d_in, const int* in_sizes, int n_in,
                              void* d_out, int out_size, void* d_ws, size_t ws_size,
                              hipStream_t stream) {
    const float* x        = (const float*)d_in[0];
    const float* ln_gamma = (const float*)d_in[1];
    const float* ln_beta  = (const float*)d_in[2];
    const float* Wq       = (const float*)d_in[3];
    const float* bq       = (const float*)d_in[4];
    const float* Wk       = (const float*)d_in[5];
    const float* bk       = (const float*)d_in[6];
    const float* Wv       = (const float*)d_in[7];
    const float* bv       = (const float*)d_in[8];
    const float* Wo       = (const float*)d_in[9];
    const float* bo       = (const float*)d_in[10];
    float* out            = (float*)d_out;

    (void)in_sizes; (void)n_in; (void)out_size;

    const size_t pack_bytes = (size_t)4 * 256 * 256 * sizeof(_Float16); // 512 KB
    const bool packed = (d_ws != nullptr && ws_size >= pack_bytes);
    _Float16* Wt = (_Float16*)d_ws;

    dim3 grid(4096);   // 4 batches * 32 * 32 windows
    dim3 block(256);   // 8 waves of 32

    if (packed) {
        dim3 pgrid(256, 4);
        dim3 pblock(256);
        pack_weights_kernel<<<pgrid, pblock, 0, stream>>>(Wq, Wk, Wv, Wo, Wt);
        win_attn_kernel<true><<<grid, block, 0, stream>>>(x, ln_gamma, ln_beta,
                                                          Wq, bq, Wk, bk, Wv, bv, Wo, bo,
                                                          Wt, out);
    } else {
        win_attn_kernel<false><<<grid, block, 0, stream>>>(x, ln_gamma, ln_beta,
                                                           Wq, bq, Wk, bk, Wv, bv, Wo, bo,
                                                           Wt, out);
    }
}
